// SelfAttention_74071005987515
// MI455X (gfx1250) — compile-verified, matrix operand order checked by reading, not run on hardware
//
#include <hip/hip_runtime.h>

// MI455X / gfx1250, wave32. bf16 WMMA (16x16x32, f32 accumulate) for all GEMMs.
// Kernel 1: x@[wq|wk|wv] + RoPE (+fold 1/sqrt(HD) into Q)  -> bf16 workspace
// Kernel 2: flash-style online-softmax attention; K tile staged via TDM
//           (tensor_load_to_lds) when available
// Kernel 3: attn_out @ wo -> f32 d_out; A tile staged via async LDS copy

#define Bb   2
#define Ss   2048
#define DIMd 4096
#define NHh  32
#define NKVv 8
#define HDd  128

typedef __bf16 bf16;
typedef __attribute__((ext_vector_type(16))) __bf16 v16bf;
typedef __attribute__((ext_vector_type(8)))  float  v8f;
typedef unsigned int u32x4 __attribute__((ext_vector_type(4)));
typedef int          i32x4 __attribute__((ext_vector_type(4)));
typedef int          i32x8 __attribute__((ext_vector_type(8)));
typedef int          vsi4  __attribute__((vector_size(4 * sizeof(int))));

#define AS1 __attribute__((address_space(1)))
#define AS3 __attribute__((address_space(3)))

// ---- CDNA5 data-mover feature detection (compile-safe on both toolchains) --
#if defined(__has_builtin)
# if __has_builtin(__builtin_amdgcn_tensor_load_to_lds)
#  define HAVE_TDM 1
# endif
# if __has_builtin(__builtin_amdgcn_global_load_async_to_lds_b128)
#  define HAVE_ASYNC_LDS 1
# endif
# if __has_builtin(__builtin_amdgcn_s_wait_asynccnt)
#  define WAIT_ASYNCCNT0() __builtin_amdgcn_s_wait_asynccnt(0)
# endif
# if __has_builtin(__builtin_amdgcn_s_wait_tensorcnt)
#  define WAIT_TENSORCNT0() __builtin_amdgcn_s_wait_tensorcnt(0)
# endif
#endif
#if defined(HAVE_ASYNC_LDS) && !defined(WAIT_ASYNCCNT0)
# define WAIT_ASYNCCNT0() asm volatile("s_wait_asynccnt 0x0" ::: "memory")
#endif
#if defined(HAVE_TDM) && !defined(WAIT_TENSORCNT0)
# define WAIT_TENSORCNT0() asm volatile("s_wait_tensorcnt 0x0" ::: "memory")
#endif

#if defined(HAVE_TDM)
# if __has_include(<hip/amd_detail/amd_gfx1250_TDM.h>)
   // amdgpu-toolchain (clang-23 / therock): 6-arg form
#  define TDM_LOAD(g0, g1) do {                                         \
     i32x4 z4_ = {}; i32x8 z8_ = {};                                    \
     __builtin_amdgcn_tensor_load_to_lds((g0), (g1), z4_, z4_, z8_, 0); \
   } while (0)
# else
   // ROCm 7.2 (clang-22): 5-arg form
#  define TDM_LOAD(g0, g1) do {                                         \
     i32x4 z4_ = {};                                                    \
     __builtin_amdgcn_tensor_load_to_lds((g0), (g1), z4_, z4_, 0);      \
   } while (0)
# endif
#endif

union FragBF { v16bf v; uint4 q[2]; };
union Pack8  { uint4 u;  bf16 h[8]; };

static __device__ inline v8f wmma_bf16(v16bf a, v16bf b, v8f c) {
  // 8 args: (neg_a, A, neg_b, B, c_mod, C, reuse_a, reuse_b)
  return __builtin_amdgcn_wmma_f32_16x16x32_bf16(false, a, false, b, (short)0, c, false, false);
}

#if defined(HAVE_ASYNC_LDS)
// 16-byte async global->LDS copy on the ASYNCcnt path.
// Builtin expects (AS1 int4*, AS3 int4*, imm offset, imm cpol).
static __device__ inline void async_b128(const bf16* gsrc, bf16* ldst) {
  __builtin_amdgcn_global_load_async_to_lds_b128(
      (AS1 vsi4*)(AS1 void*)(void*)gsrc,
      (AS3 vsi4*)(AS3 void*)(void*)ldst, 0, 0);
}
#endif

// ---------------------------------------------------------------------------
// Kernel 1: QKV projection + RoPE.  M = B*S = 4096 rows of x, N = 6144 cols
// (q: 0..4095 -> wq, k: 4096..5119 -> wk, v: 5120..6143 -> wv).
// Block: 256 threads = 8 waves, 64x64 output tile; each wave does 16x32.
// ---------------------------------------------------------------------------
__global__ __launch_bounds__(256)
void qkv_rope_kernel(const float* __restrict__ x,
                     const float* __restrict__ wq,
                     const float* __restrict__ wk,
                     const float* __restrict__ wv,
                     const float* __restrict__ fcos,
                     const float* __restrict__ fsin,
                     bf16* __restrict__ qws,
                     bf16* __restrict__ kws,
                     bf16* __restrict__ vws)
{
  __shared__ __align__(16) bf16  lds_a [64 * 32];   // A tile, [row][k]
  __shared__ __align__(16) bf16  lds_bt[64 * 32];   // B tile transposed, [n][k]
  __shared__ __align__(16) float lds_c [64 * 64];   // f32 C tile for RoPE epilogue

  const int tid  = threadIdx.x;
  const int lane = tid & 31;
  const int w    = tid >> 5;
  const int ln   = lane & 15, lh = lane >> 4;
  const int wm   = w & 3,     wn = w >> 2;
  const int mbase = blockIdx.x * 64;
  const int nbase = blockIdx.y * 64;

  const float* wptr; int ldn, col0;
  if (nbase < 4096)      { wptr = wq; ldn = 4096; col0 = nbase;        }
  else if (nbase < 5120) { wptr = wk; ldn = 1024; col0 = nbase - 4096; }
  else                   { wptr = wv; ldn = 1024; col0 = nbase - 5120; }

  v8f acc0 = {}, acc1 = {};

  for (int kk = 0; kk < DIMd; kk += 32) {
    // prefetch next K-step tiles toward L2 while this step computes
    if (kk + 32 < DIMd) {
      __builtin_prefetch(x + (size_t)(mbase + (tid >> 3)) * DIMd + kk + 32 + (tid & 7) * 4, 0, 0);
      __builtin_prefetch(wptr + (size_t)(kk + 32 + (tid >> 4)) * ldn + col0 + (tid & 15) * 4, 0, 0);
    }
    // stage A: 64x32 f32 -> bf16 (512 float4, 2 per thread)
#pragma unroll
    for (int i = 0; i < 2; ++i) {
      int id  = tid + 256 * i;
      int row = id >> 3, c4 = (id & 7) * 4;
      float4 v = *(const float4*)(x + (size_t)(mbase + row) * DIMd + kk + c4);
      lds_a[row * 32 + c4 + 0] = (bf16)v.x;
      lds_a[row * 32 + c4 + 1] = (bf16)v.y;
      lds_a[row * 32 + c4 + 2] = (bf16)v.z;
      lds_a[row * 32 + c4 + 3] = (bf16)v.w;
    }
    // stage B transposed: W is [k][n] row-major -> LDS [n][k]
#pragma unroll
    for (int i = 0; i < 2; ++i) {
      int id = tid + 256 * i;
      int k  = id >> 4, n4 = (id & 15) * 4;
      float4 v = *(const float4*)(wptr + (size_t)(kk + k) * ldn + col0 + n4);
      lds_bt[(n4 + 0) * 32 + k] = (bf16)v.x;
      lds_bt[(n4 + 1) * 32 + k] = (bf16)v.y;
      lds_bt[(n4 + 2) * 32 + k] = (bf16)v.z;
      lds_bt[(n4 + 3) * 32 + k] = (bf16)v.w;
    }
    __syncthreads();

    // fragments: per-lane K striping {kb..kb+7, 16+kb..16+kb+7}, kb=(lane/16)*8
    FragBF a, b0, b1;
    const bf16* ar = lds_a + (wm * 16 + ln) * 32 + lh * 8;
    a.q[0] = *(const uint4*)(ar);
    a.q[1] = *(const uint4*)(ar + 16);
    const bf16* b0r = lds_bt + (wn * 32 + ln) * 32 + lh * 8;
    b0.q[0] = *(const uint4*)(b0r);
    b0.q[1] = *(const uint4*)(b0r + 16);
    const bf16* b1r = lds_bt + (wn * 32 + 16 + ln) * 32 + lh * 8;
    b1.q[0] = *(const uint4*)(b1r);
    b1.q[1] = *(const uint4*)(b1r + 16);

    acc0 = wmma_bf16(a.v, b0.v, acc0);
    acc1 = wmma_bf16(a.v, b1.v, acc1);
    __syncthreads();
  }

  // C layout: lane ln holds column, VGPR r holds row r + 8*(lane/16)
#pragma unroll
  for (int r = 0; r < 8; ++r) {
    lds_c[(wm * 16 + r + 8 * lh) * 64 + wn * 32 + ln]      = acc0[r];
    lds_c[(wm * 16 + r + 8 * lh) * 64 + wn * 32 + 16 + ln] = acc1[r];
  }
  __syncthreads();

  // epilogue: RoPE on (even,odd) column pairs, 2048 pairs / 256 threads
  const float qscale = 0.08838834764831845f;  // 1/sqrt(128), folded into Q
#pragma unroll
  for (int i = 0; i < 8; ++i) {
    int pid = tid + 256 * i;
    int row = pid >> 5, pc = pid & 31;
    int m = mbase + row;
    int n = nbase + pc * 2;
    int s = m & (Ss - 1);
    float e0 = lds_c[row * 64 + pc * 2];
    float e1 = lds_c[row * 64 + pc * 2 + 1];
    if (nbase < 4096) {                    // Q: rope + scale
      int p = (n & 127) >> 1;
      float c = fcos[s * 64 + p], sn = fsin[s * 64 + p];
      qws[(size_t)m * 4096 + n]     = (bf16)((e0 * c - e1 * sn) * qscale);
      qws[(size_t)m * 4096 + n + 1] = (bf16)((e0 * sn + e1 * c) * qscale);
    } else if (nbase < 5120) {             // K: rope
      int nk = n - 4096;
      int p = (nk & 127) >> 1;
      float c = fcos[s * 64 + p], sn = fsin[s * 64 + p];
      kws[(size_t)m * 1024 + nk]     = (bf16)(e0 * c - e1 * sn);
      kws[(size_t)m * 1024 + nk + 1] = (bf16)(e0 * sn + e1 * c);
    } else {                               // V: plain
      int nv = n - 5120;
      vws[(size_t)m * 1024 + nv]     = (bf16)e0;
      vws[(size_t)m * 1024 + nv + 1] = (bf16)e1;
    }
  }
}

// ---------------------------------------------------------------------------
// Kernel 2: attention. Grid (S/64, NH, B); block = 128 threads = 4 waves,
// each wave owns 16 query rows. Loop over 32-key tiles with online softmax.
// K tile staged by the Tensor Data Mover (one descriptor per tile) when the
// toolchain exposes the builtin; V tile transposed manually into LDS.
// ---------------------------------------------------------------------------
__global__ __launch_bounds__(128)
void attn_kernel(const bf16* __restrict__ qws,
                 const bf16* __restrict__ kws,
                 const bf16* __restrict__ vws,
                 bf16* __restrict__ ows)
{
  __shared__ __align__(16) bf16 lds_k [32 * 128];   // K tile [key][d]
  __shared__ __align__(16) bf16 lds_vt[128 * 32];   // V tile transposed [d][key]
  __shared__ __align__(16) bf16 lds_p [4 * 16 * 32];// per-wave P relayout patch

  const int tid  = threadIdx.x;
  const int lane = tid & 31;
  const int w    = tid >> 5;
  const int ln   = lane & 15, lh = lane >> 4;
  const int qt = blockIdx.x, h = blockIdx.y, b = blockIdx.z;
  const int kvh = h >> 2;                 // N_REP = 4

  // Q fragments: 16 rows x 128 dims -> 4 chunks of 16x32
  FragBF qf[4];
  {
    const int qrow = qt * 64 + w * 16 + ln;
    const bf16* qb = qws + ((size_t)(b * Ss + qrow)) * 4096 + h * HDd;
#pragma unroll
    for (int c = 0; c < 4; ++c) {
      qf[c].q[0] = *(const uint4*)(qb + c * 32 + lh * 8);
      qf[c].q[1] = *(const uint4*)(qb + c * 32 + 16 + lh * 8);
    }
  }

  v8f o[8] = {};                          // 16x128 f32 output accumulator
  float mrow[8], lrow[8];
#pragma unroll
  for (int r = 0; r < 8; ++r) { mrow[r] = -1e30f; lrow[r] = 0.f; }

  for (int kt = 0; kt < Ss / 32; ++kt) {
    __syncthreads();                      // protect K/V/P vs previous readers

#if defined(HAVE_TDM)
    // ---- TDM: one 2-D descriptor pulls the 32x128 bf16 K tile into LDS ----
    if (w == 0) {
      size_t gelem = ((size_t)(b * Ss + kt * 32) * NKVv + kvh) * HDd;
      unsigned long long ga = (unsigned long long)(const void*)(kws + gelem);
      unsigned lds_off = (unsigned)(size_t)(AS3 char*)&lds_k[0];
      u32x4 g0;
      g0[0] = 1u;                                        // count=1, user mode
      g0[1] = lds_off;                                   // lds_addr (bytes)
      g0[2] = (unsigned)(ga & 0xffffffffu);              // global_addr[31:0]
      g0[3] = (unsigned)((ga >> 32) & 0x1ffffffu) | (2u << 30); // ga[56:32] | type=2
      i32x8 g1;
      g1[0] = (int)(1u << 16);          // data_size=1 (2 bytes/elem)
      g1[1] = (int)(128u << 16);        // tensor_dim0 = 128 (elements per row)
      g1[2] = (int)(32u  << 16);        // tensor_dim1 = 32 rows
      g1[3] = (int)(128u << 16);        // tile_dim0   = 128
      g1[4] = 32;                       // tile_dim1   = 32, tile_dim2 = 0
      g1[5] = 1024;                     // tensor_dim0_stride = NKV*HD elements
      g1[6] = 0;
      g1[7] = 0;
      TDM_LOAD(g0, g1);
    }
#endif

    // stage V transposed (and K manually when TDM is unavailable)
#pragma unroll
    for (int i = 0; i < 4; ++i) {
      int id  = tid + 128 * i;
      int key = id >> 4, d8 = (id & 15) * 8;
      size_t g = ((size_t)(b * Ss + kt * 32 + key) * NKVv + kvh) * HDd + d8;
#if !defined(HAVE_TDM)
      *(uint4*)(lds_k + key * 128 + d8) = *(const uint4*)(kws + g);
#endif
      Pack8 p8; p8.u = *(const uint4*)(vws + g);
#pragma unroll
      for (int j = 0; j < 8; ++j) lds_vt[(d8 + j) * 32 + key] = p8.h[j];
    }
#if defined(HAVE_TDM)
    if (w == 0) { WAIT_TENSORCNT0(); }    // DMA complete before publishing
#endif
    __syncthreads();

    // S = Q @ K^T : two 16-key accumulators, 4 d-chunks each
    v8f s0 = {}, s1 = {};
#pragma unroll
    for (int c = 0; c < 4; ++c) {
      FragBF k0, k1;
      const bf16* k0r = lds_k + ln * 128 + c * 32 + lh * 8;
      k0.q[0] = *(const uint4*)(k0r);
      k0.q[1] = *(const uint4*)(k0r + 16);
      const bf16* k1r = lds_k + (16 + ln) * 128 + c * 32 + lh * 8;
      k1.q[0] = *(const uint4*)(k1r);
      k1.q[1] = *(const uint4*)(k1r + 16);
      s0 = wmma_bf16(qf[c].v, k0.v, s0);
      s1 = wmma_bf16(qf[c].v, k1.v, s1);
    }

    // online softmax; row stats reduced over the 16 lanes of each half-wave
    bf16* pw = lds_p + w * 512;
#pragma unroll
    for (int r = 0; r < 8; ++r) {
      float a = s0[r], c2 = s1[r];
      float mx = fmaxf(a, c2);
#pragma unroll
      for (int m2 = 8; m2 >= 1; m2 >>= 1) mx = fmaxf(mx, __shfl_xor(mx, m2, 32));
      float nm   = fmaxf(mrow[r], mx);
      float p0   = __expf(a  - nm);
      float p1   = __expf(c2 - nm);
      float corr = __expf(mrow[r] - nm);
      float ps = p0 + p1;
#pragma unroll
      for (int m2 = 8; m2 >= 1; m2 >>= 1) ps += __shfl_xor(ps, m2, 32);
      lrow[r] = lrow[r] * corr + ps;
      mrow[r] = nm;
#pragma unroll
      for (int t2 = 0; t2 < 8; ++t2) o[t2][r] = o[t2][r] * corr;
      // C-layout -> LDS [row][key] so it can be re-read in A-layout
      pw[(r + 8 * lh) * 32 + ln]      = (bf16)p0;
      pw[(r + 8 * lh) * 32 + 16 + ln] = (bf16)p1;
    }
    __syncthreads();

    // O += P @ V : A = P (16x32 keys), 8 d-tiles of V
    FragBF pf;
    pf.q[0] = *(const uint4*)(pw + ln * 32 + lh * 8);
    pf.q[1] = *(const uint4*)(pw + ln * 32 + 16 + lh * 8);
#pragma unroll
    for (int dt = 0; dt < 8; ++dt) {
      FragBF vf;
      const bf16* vr = lds_vt + (dt * 16 + ln) * 32 + lh * 8;
      vf.q[0] = *(const uint4*)(vr);
      vf.q[1] = *(const uint4*)(vr + 16);
      o[dt] = wmma_bf16(pf.v, vf.v, o[dt]);
    }
  }

  // normalize and store bf16 attention output (b, s, h, d)
#pragma unroll
  for (int r = 0; r < 8; ++r) {
    float inv  = 1.0f / lrow[r];
    int   qrow = qt * 64 + w * 16 + r + 8 * lh;
    size_t base = ((size_t)(b * Ss + qrow) * NHh + h) * HDd;
#pragma unroll
    for (int dt = 0; dt < 8; ++dt)
      ows[base + dt * 16 + ln] = (bf16)(o[dt][r] * inv);
  }
}

// ---------------------------------------------------------------------------
// Kernel 3: out = attn_out @ wo  (4096 x 4096 x 4096), f32 output.
// A tile is already bf16 -> staged with global_load_async_to_lds_b128
// (ASYNCcnt path) when available.
// ---------------------------------------------------------------------------
__global__ __launch_bounds__(256)
void oproj_kernel(const bf16* __restrict__ aws,
                  const float* __restrict__ wo,
                  float* __restrict__ out)
{
  __shared__ __align__(16) bf16 lds_a [64 * 32];
  __shared__ __align__(16) bf16 lds_bt[64 * 32];

  const int tid  = threadIdx.x;
  const int lane = tid & 31;
  const int w    = tid >> 5;
  const int ln   = lane & 15, lh = lane >> 4;
  const int wm   = w & 3,     wn = w >> 2;
  const int mbase = blockIdx.x * 64;
  const int nbase = blockIdx.y * 64;

  v8f acc0 = {}, acc1 = {};

  for (int kk = 0; kk < 4096; kk += 32) {
    if (kk + 32 < 4096)
      __builtin_prefetch(wo + (size_t)(kk + 32 + (tid >> 4)) * 4096 + nbase + (tid & 15) * 4, 0, 0);

    {  // A already bf16: 256 x 16B, 1 chunk per thread
      int row = tid >> 2, c8 = (tid & 3) * 8;
      const bf16* gsrc = aws + (size_t)(mbase + row) * 4096 + kk + c8;
      bf16*       ldst = lds_a + row * 32 + c8;
#if defined(HAVE_ASYNC_LDS)
      async_b128(gsrc, ldst);
#else
      *(uint4*)ldst = *(const uint4*)gsrc;
#endif
    }
#pragma unroll
    for (int i = 0; i < 2; ++i) {
      int id = tid + 256 * i;
      int k  = id >> 4, n4 = (id & 15) * 4;
      float4 v = *(const float4*)(wo + (size_t)(kk + k) * 4096 + nbase + n4);
      lds_bt[(n4 + 0) * 32 + k] = (bf16)v.x;
      lds_bt[(n4 + 1) * 32 + k] = (bf16)v.y;
      lds_bt[(n4 + 2) * 32 + k] = (bf16)v.z;
      lds_bt[(n4 + 3) * 32 + k] = (bf16)v.w;
    }
#if defined(HAVE_ASYNC_LDS)
    WAIT_ASYNCCNT0();
#endif
    __syncthreads();

    FragBF a, b0, b1;
    const bf16* ar = lds_a + (wm * 16 + ln) * 32 + lh * 8;
    a.q[0] = *(const uint4*)(ar);
    a.q[1] = *(const uint4*)(ar + 16);
    const bf16* b0r = lds_bt + (wn * 32 + ln) * 32 + lh * 8;
    b0.q[0] = *(const uint4*)(b0r);
    b0.q[1] = *(const uint4*)(b0r + 16);
    const bf16* b1r = lds_bt + (wn * 32 + 16 + ln) * 32 + lh * 8;
    b1.q[0] = *(const uint4*)(b1r);
    b1.q[1] = *(const uint4*)(b1r + 16);

    acc0 = wmma_bf16(a.v, b0.v, acc0);
    acc1 = wmma_bf16(a.v, b1.v, acc1);
    __syncthreads();
  }

#pragma unroll
  for (int r = 0; r < 8; ++r) {
    int m = mbase + wm * 16 + r + 8 * lh;
    out[(size_t)m * 4096 + nbase + wn * 32 + ln]      = acc0[r];
    out[(size_t)m * 4096 + nbase + wn * 32 + 16 + ln] = acc1[r];
  }
}

// ---------------------------------------------------------------------------
extern "C" void kernel_launch(void* const* d_in, const int* in_sizes, int n_in,
                              void* d_out, int out_size, void* d_ws, size_t ws_size,
                              hipStream_t stream)
{
  (void)in_sizes; (void)n_in; (void)out_size; (void)ws_size;
  const float* x  = (const float*)d_in[0];
  const float* wq = (const float*)d_in[1];
  const float* wk = (const float*)d_in[2];
  const float* wv = (const float*)d_in[3];
  const float* wo = (const float*)d_in[4];
  const float* fc = (const float*)d_in[5];
  const float* fs = (const float*)d_in[6];
  // d_in[7..9]: cache_k, cache_v, start_pos(=0) — with start_pos==0 and s==S
  // the updated cache equals the freshly computed K/V, so they are not read.

  char* ws = (char*)d_ws;
  const size_t QB = (size_t)4096 * 4096 * 2;  // q bf16
  const size_t KB = (size_t)4096 * 1024 * 2;  // k bf16
  bf16* qws = (bf16*)(ws);
  bf16* kws = (bf16*)(ws + QB);
  bf16* vws = (bf16*)(ws + QB + KB);
  bf16* aws = (bf16*)(ws + QB + 2 * KB);

  qkv_rope_kernel<<<dim3(64, 96), dim3(256), 0, stream>>>(x, wq, wk, wv, fc, fs,
                                                          qws, kws, vws);
  attn_kernel<<<dim3(32, 32, 2), dim3(128), 0, stream>>>(qws, kws, vws, aws);
  oproj_kernel<<<dim3(64, 64), dim3(256), 0, stream>>>(aws, wo, (float*)d_out);
}
